// Attention_45930380263513
// MI455X (gfx1250) — matrix operand order repaired and checked by reading, not compiled, not run on hardware
//
#include <hip/hip_runtime.h>

// ---------------- problem constants ----------------
#define DIMX      512
#define KEY_DIMX  32
#define NUM_HEADS 16
#define VAL_DIMX  128
#define OUT_DIMX  (VAL_DIMX * NUM_HEADS)       // 2048
#define NTOK      1024
#define BATCH     16
#define PER_HEAD  (2 * KEY_DIMX + VAL_DIMX)    // 192
#define QKV_OUT   (NUM_HEADS * PER_HEAD)       // 3072
#define SCALEQK   0.17677669529663687f         // 32^-0.5
#define LOG2E     1.4426950408889634f
#define LN_EPS    1e-5f

typedef __attribute__((ext_vector_type(4)))  unsigned int u32x4;
typedef __attribute__((ext_vector_type(4)))  int          i32x4;
typedef __attribute__((ext_vector_type(8)))  int          i32x8;
typedef __attribute__((ext_vector_type(4)))  float        f32x4;
typedef __attribute__((ext_vector_type(16))) __bf16       v16bf;
typedef __attribute__((ext_vector_type(8)))  float        v8f;

union Frag  { v16bf v; u32x4 q[2]; unsigned short h[16]; };
union CFrag { v8f   v; f32x4 q[2]; };

// hardware RNE f32->bf16 (lowers to v_cvt_pk_bf16_f32 family)
__device__ __forceinline__ unsigned short f2bf(float f) {
    union { __bf16 b; unsigned short u; } c;
    c.b = (__bf16)f;
    return c.u;
}

// ---- DPP16 butterfly reductions over the 16-lane half-wave ----
// DPP applied as a source modifier on the combining op itself: 1 VALU per step
// (avoids the mov_dpp + canonicalize + max triple clang emits for shuffles).
__device__ __forceinline__ float red_max16(float x) {
    float y;
    asm("v_max_num_f32_dpp %0, %1, %1 quad_perm:[1,0,3,2] row_mask:0xf bank_mask:0xf bound_ctrl:1"
        : "=&v"(y) : "v"(x));
    asm("v_max_num_f32_dpp %0, %1, %1 quad_perm:[2,3,0,1] row_mask:0xf bank_mask:0xf bound_ctrl:1"
        : "=&v"(x) : "v"(y));
    asm("v_max_num_f32_dpp %0, %1, %1 row_half_mirror row_mask:0xf bank_mask:0xf bound_ctrl:1"
        : "=&v"(y) : "v"(x));
    asm("v_max_num_f32_dpp %0, %1, %1 row_mirror row_mask:0xf bank_mask:0xf bound_ctrl:1"
        : "=&v"(x) : "v"(y));
    return x;
}
__device__ __forceinline__ float red_sum16(float x) {
    float y;
    asm("v_add_f32_dpp %0, %1, %1 quad_perm:[1,0,3,2] row_mask:0xf bank_mask:0xf bound_ctrl:1"
        : "=&v"(y) : "v"(x));
    asm("v_add_f32_dpp %0, %1, %1 quad_perm:[2,3,0,1] row_mask:0xf bank_mask:0xf bound_ctrl:1"
        : "=&v"(x) : "v"(y));
    asm("v_add_f32_dpp %0, %1, %1 row_half_mirror row_mask:0xf bank_mask:0xf bound_ctrl:1"
        : "=&v"(y) : "v"(x));
    asm("v_add_f32_dpp %0, %1, %1 row_mirror row_mask:0xf bank_mask:0xf bound_ctrl:1"
        : "=&v"(x) : "v"(y));
    return x;
}

// ---- Tensor Data Mover: DMA a [256 rows x 32 cols] bf16 tile of W[N][K] into LDS ----
// D# packed per CDNA5 ISA ch.8: group0 = {count, lds_addr, global_addr, type=2},
// group1 = {data_size=2B, tensor dims, tile dims 32x256, row stride K}.
// Rows land contiguously (64B each) == Wt[256][32] row-major layout.
// This toolchain's builtin takes 6 args (extra int32x8 group before cpol).
__device__ __forceinline__ void tdm_load_w(const unsigned short* gptr, unsigned lds_off,
                                           int K, int N) {
    unsigned long long ga = (unsigned long long)(size_t)gptr;  // byte address
    u32x4 g0;
    g0[0] = 1u;                                            // count=1, valid
    g0[1] = lds_off;                                       // LDS byte address
    g0[2] = (unsigned)(ga & 0xffffffffull);                // global_addr[31:0]
    g0[3] = (unsigned)((ga >> 32) & 0x1ffffffull) | (2u << 30);  // addr[56:32] | type=2
    i32x8 g1;
    g1[0] = (int)(1u << 16);                 // workgroup_mask=0, data_size=1 (2 bytes)
    g1[1] = (int)(((unsigned)K & 0xffffu) << 16);  // tensor_dim0[15:0] (bits 63:48)
    g1[2] = (int)(((unsigned)N & 0xffffu) << 16);  // dim0[31:16]=0 | tensor_dim1[15:0]
    g1[3] = (int)(32u << 16);                // dim1[31:16]=0 | tile_dim0=32
    g1[4] = 256;                             // tile_dim1=256 | tile_dim2=0
    g1[5] = K;                               // tensor_dim0_stride[31:0] (elements)
    g1[6] = 0;                               // stride0 hi | stride1 lo
    g1[7] = 0;
    i32x4 z4 = {0, 0, 0, 0};
    i32x8 z8 = {0, 0, 0, 0, 0, 0, 0, 0};
    __builtin_amdgcn_tensor_load_to_lds(g0, g1, z4, z4, z8, 0);
}

// ---------------- fp32 -> bf16 weight convert ----------------
__global__ __launch_bounds__(256) void cvt_kernel(const float* __restrict__ in,
                                                  unsigned short* __restrict__ out, int n) {
    int i = blockIdx.x * 256 + threadIdx.x;
    if (i < n) out[i] = f2bf(in[i]);
}

// ---------------- LayerNorm (one row per block) -> bf16 ----------------
__global__ __launch_bounds__(256) void ln_kernel(const float* __restrict__ x,
                                                 const float* __restrict__ gamma,
                                                 const float* __restrict__ beta,
                                                 unsigned short* __restrict__ xn) {
    __shared__ float rs[256], rs2[256];
    const int row = blockIdx.x;
    const int t = threadIdx.x;
    const float* xr = x + (size_t)row * DIMX;
    float a = xr[t], b = xr[t + 256];
    rs[t] = a + b;
    rs2[t] = a * a + b * b;
    __syncthreads();
    for (int off = 128; off > 0; off >>= 1) {
        if (t < off) { rs[t] += rs[t + off]; rs2[t] += rs2[t + off]; }
        __syncthreads();
    }
    float mu  = rs[0] * (1.0f / DIMX);
    float var = rs2[0] * (1.0f / DIMX) - mu * mu;
    float inv = rsqrtf(var + LN_EPS);
    unsigned short* o = xn + (size_t)row * DIMX;
    o[t]       = f2bf((a - mu) * inv * gamma[t]       + beta[t]);
    o[t + 256] = f2bf((b - mu) * inv * gamma[t + 256] + beta[t + 256]);
}

// -------- bias expansion, TRANSPOSED + log2e-scaled: bm[h][k][q] = ab[h][idx[q][k]]*log2e --
__global__ __launch_bounds__(256) void bias_kernel(const float* __restrict__ ab,
                                                   const int* __restrict__ idx,
                                                   float* __restrict__ bm) {
    int i = blockIdx.x * 256 + threadIdx.x;   // 16 * 1024 * 1024 total
    int h = i >> 20;
    int k = (i >> 10) & 1023;
    int q = i & 1023;
    bm[i] = ab[(h << 10) + idx[q * NTOK + k]] * LOG2E;  // idx (4MB) stays L2-resident
}

// ---------------- tiled bf16 WMMA GEMM: C[m][n] = sum_k A[m][k]*W[n][k] + bias[n] ----
// Block tile 64(M) x 256(N), 8 waves in 2x4 grid, each wave 32x64 (2x4 16x16 frags).
// W tile staged by the Tensor Data Mover (wave 0 issues, s_wait_tensorcnt gates the
// barrier); A tile register-double-buffered. One barrier per K-step; next tile's
// TDM/global loads overlap the WMMAs.
template <int OUT_F32, int QSCALE>
__global__ __launch_bounds__(256) void gemm_kernel(const unsigned short* __restrict__ A,
                                                   const unsigned short* __restrict__ W,
                                                   const float* __restrict__ bias,
                                                   void* __restrict__ Cout,
                                                   int M, int N, int K) {
    __shared__ unsigned short At[2][64 * 32];
    __shared__ unsigned short Wt[2][256 * 32];
    const int t    = threadIdx.x;
    const int lane = t & 31, wid = t >> 5;
    const int wm = wid & 1, wn = wid >> 1;
    const int mb = blockIdx.y * 64;
    const int nb = blockIdx.x * 256;
    const int l16 = lane & 15, hi = lane >> 4;

    v8f acc[2][4];
#pragma unroll
    for (int i = 0; i < 2; i++)
#pragma unroll
        for (int j = 0; j < 4; j++) acc[i][j] = (v8f){0.f, 0.f, 0.f, 0.f, 0.f, 0.f, 0.f, 0.f};

    const int ar = t >> 2, ak = (t & 3) * 8;  // A staging: 8 bf16 (16B) per thread
    const unsigned short* wtile = W + (size_t)nb * K;

    if (wid == 0) tdm_load_w(wtile, (unsigned)(size_t)&Wt[0][0], K, N);  // tile 0
    u32x4 ra = *(const u32x4*)&A[(size_t)(mb + ar) * K + ak];

    int buf = 0;
    for (int kb = 0; kb < K; kb += 32, buf ^= 1) {
        *(u32x4*)&At[buf][ar * 32 + ak] = ra;
        if (wid == 0) __builtin_amdgcn_s_wait_tensorcnt(0);  // current W tile landed
        __syncthreads();

        if (kb + 32 < K) {  // kick off next tile (overlaps with WMMA below)
            if (wid == 0)
                tdm_load_w(wtile + kb + 32, (unsigned)(size_t)&Wt[buf ^ 1][0], K, N);
            ra = *(const u32x4*)&A[(size_t)(mb + ar) * K + kb + 32 + ak];
        }

        Frag a[2], b[4];
#pragma unroll
        for (int i = 0; i < 2; i++) {
            const unsigned short* pa = &At[buf][(wm * 32 + i * 16 + l16) * 32];
            a[i].q[0] = *(const u32x4*)&pa[hi * 8];
            a[i].q[1] = *(const u32x4*)&pa[16 + hi * 8];
        }
#pragma unroll
        for (int j = 0; j < 4; j++) {
            const unsigned short* pw = &Wt[buf][(wn * 64 + j * 16 + l16) * 32];
            b[j].q[0] = *(const u32x4*)&pw[hi * 8];
            b[j].q[1] = *(const u32x4*)&pw[16 + hi * 8];
        }
#pragma unroll
        for (int i = 0; i < 2; i++)
#pragma unroll
            for (int j = 0; j < 4; j++)
                acc[i][j] = __builtin_amdgcn_wmma_f32_16x16x32_bf16(
                    false, a[i].v, false, b[j].v, (short)0, acc[i][j], false, false);
    }

#pragma unroll
    for (int i = 0; i < 2; i++)
#pragma unroll
        for (int j = 0; j < 4; j++) {
            int ncol = nb + wn * 64 + j * 16 + l16;
            float bv = bias[ncol];
            float qs = 1.0f;
            if (QSCALE && (ncol % PER_HEAD) < KEY_DIMX) qs = SCALEQK * LOG2E;
#pragma unroll
            for (int r = 0; r < 8; r++) {
                int mrow = mb + wm * 32 + i * 16 + r + 8 * hi;
                float v = (acc[i][j][r] + bv) * qs;
                if (OUT_F32) ((float*)Cout)[(size_t)mrow * N + ncol] = v;
                else ((unsigned short*)Cout)[(size_t)mrow * N + ncol] = f2bf(v);
            }
        }
}

// ---------------- flash attention: block = (qtile of 64, head, batch); 4 waves ----------------
__global__ __launch_bounds__(128) void attn_kernel(const unsigned short* __restrict__ qkv,
                                                   const float* __restrict__ bm,
                                                   unsigned short* __restrict__ out) {
    __shared__ unsigned short Vt[2][128 * 40];  // V transposed: [v][key], stride 40
    __shared__ unsigned short Kt[2][32 * 32];   // K chunk row-major [key][k]
    __shared__ unsigned short Pt[4][16 * 32];   // per-wave P staging tile
    const int t = threadIdx.x, lane = t & 31, wid = t >> 5;
    const int l16 = lane & 15, hi = lane >> 4;
    const int qt = blockIdx.x, h = blockIdx.y, b = blockIdx.z;
    const size_t headoff = (size_t)h * PER_HEAD;
    const int q0 = qt * 64 + wid * 16;

    // Q A-fragment (16 rows x K=32), loaded once (scale*log2e folded in by GEMM1)
    Frag qf;
    {
        const unsigned short* p = qkv + ((size_t)(b * NTOK + q0 + l16)) * QKV_OUT + headoff;
        qf.q[0] = *(const u32x4*)&p[hi * 8];
        qf.q[1] = *(const u32x4*)&p[16 + hi * 8];
    }

    v8f acc[8];
#pragma unroll
    for (int f = 0; f < 8; f++) acc[f] = (v8f){0.f, 0.f, 0.f, 0.f, 0.f, 0.f, 0.f, 0.f};
    float m_run[8], l_run[8];
#pragma unroll
    for (int r = 0; r < 8; r++) { m_run[r] = -1e30f; l_run[r] = 0.f; }

    const int skey = t & 31;                    // staging: key within chunk
    const int sgrp = t >> 5;                    // staging group 0..3
    const float* bmh = bm + ((size_t)h << 20);  // bm[h][k][q]

    // preload chunk 0 into registers
    u32x4 ku;
    Frag va, vb;
    {
        const unsigned short* kp =
            qkv + ((size_t)(b * NTOK + skey)) * QKV_OUT + headoff + KEY_DIMX + sgrp * 8;
        ku = *(const u32x4*)kp;
        const unsigned short* vp =
            qkv + ((size_t)(b * NTOK + skey)) * QKV_OUT + headoff + 2 * KEY_DIMX + sgrp * 32;
        va.q[0] = *(const u32x4*)&vp[0];  va.q[1] = *(const u32x4*)&vp[8];
        vb.q[0] = *(const u32x4*)&vp[16]; vb.q[1] = *(const u32x4*)&vp[24];
    }

    int buf = 0;
    for (int ch = 0; ch < NTOK; ch += 32, buf ^= 1) {
        // stage current chunk from registers
        *(u32x4*)&Kt[buf][skey * 32 + sgrp * 8] = ku;
#pragma unroll
        for (int v = 0; v < 16; v++) Vt[buf][(sgrp * 32 + v) * 40 + skey]      = va.h[v];
#pragma unroll
        for (int v = 0; v < 16; v++) Vt[buf][(sgrp * 32 + 16 + v) * 40 + skey] = vb.h[v];
        __syncthreads();

        if (ch + 32 < NTOK) {  // preload next chunk (overlaps compute)
            const unsigned short* kp =
                qkv + ((size_t)(b * NTOK + ch + 32 + skey)) * QKV_OUT + headoff + KEY_DIMX + sgrp * 8;
            ku = *(const u32x4*)kp;
            const unsigned short* vp =
                qkv + ((size_t)(b * NTOK + ch + 32 + skey)) * QKV_OUT + headoff + 2 * KEY_DIMX + sgrp * 32;
            va.q[0] = *(const u32x4*)&vp[0];  va.q[1] = *(const u32x4*)&vp[8];
            vb.q[0] = *(const u32x4*)&vp[16]; vb.q[1] = *(const u32x4*)&vp[24];
        }

        // K B-fragments from LDS
        Frag kf0, kf1;
        {
            const unsigned short* p0 = &Kt[buf][(l16) * 32 + hi * 8];
            const unsigned short* p1 = &Kt[buf][(16 + l16) * 32 + hi * 8];
            kf0.q[0] = *(const u32x4*)&p0[0]; kf0.q[1] = *(const u32x4*)&p0[16];
            kf1.q[0] = *(const u32x4*)&p1[0]; kf1.q[1] = *(const u32x4*)&p1[16];
        }
        // bias tiles load directly into the WMMA C operand (bm[h][k][q] layout)
        CFrag c0, c1;
        {
            const float* b0 = &bmh[(size_t)(ch + l16) * NTOK + q0 + hi * 8];
            const float* b1 = &bmh[(size_t)(ch + 16 + l16) * NTOK + q0 + hi * 8];
            c0.q[0] = *(const f32x4*)&b0[0]; c0.q[1] = *(const f32x4*)&b0[4];
            c1.q[0] = *(const f32x4*)&b1[0]; c1.q[1] = *(const f32x4*)&b1[4];
        }
        v8f s0 = __builtin_amdgcn_wmma_f32_16x16x32_bf16(false, qf.v, false, kf0.v, (short)0, c0.v, false, false);
        v8f s1 = __builtin_amdgcn_wmma_f32_16x16x32_bf16(false, qf.v, false, kf1.v, (short)0, c1.v, false, false);

        // online softmax in log2 domain; row reductions via fused DPP butterflies
        float p0v[8], p1v[8];
#pragma unroll
        for (int r = 0; r < 8; r++) {
            float v0 = s0[r], v1 = s1[r];
            float mnew = fmaxf(m_run[r], red_max16(fmaxf(v0, v1)));
            v0 = exp2f(v0 - mnew);
            v1 = exp2f(v1 - mnew);
            float rsum = red_sum16(v0 + v1);
            float corr = exp2f(m_run[r] - mnew);
            l_run[r] = l_run[r] * corr + rsum;
            m_run[r] = mnew;
            p0v[r] = v0; p1v[r] = v1;
#pragma unroll
            for (int f = 0; f < 8; f++) acc[f][r] *= corr;
        }

        // C-layout P -> A-fragment layout via per-wave LDS tile (wave-local ordering only)
        unsigned short* pt = Pt[wid];
#pragma unroll
        for (int r = 0; r < 8; r++) {
            pt[(r + 8 * hi) * 32 + l16]      = f2bf(p0v[r]);
            pt[(r + 8 * hi) * 32 + 16 + l16] = f2bf(p1v[r]);
        }
        __builtin_amdgcn_wave_barrier();  // wave-local fence; hw dscnt waits enforce RAW
        Frag pf;
        pf.q[0] = *(const u32x4*)&pt[l16 * 32 + hi * 8];
        pf.q[1] = *(const u32x4*)&pt[l16 * 32 + 16 + hi * 8];

        // O += P @ V  (8 fragments cover v-dim 128)
#pragma unroll
        for (int f = 0; f < 8; f++) {
            Frag vf;
            const unsigned short* vp = &Vt[buf][(f * 16 + l16) * 40 + hi * 8];
            vf.q[0] = *(const u32x4*)&vp[0];
            vf.q[1] = *(const u32x4*)&vp[16];
            acc[f] = __builtin_amdgcn_wmma_f32_16x16x32_bf16(
                false, pf.v, false, vf.v, (short)0, acc[f], false, false);
        }
    }

    float inv[8];
#pragma unroll
    for (int r = 0; r < 8; r++) inv[r] = 1.0f / l_run[r];
#pragma unroll
    for (int f = 0; f < 8; f++)
#pragma unroll
        for (int r = 0; r < 8; r++) {
            int qrow = q0 + r + 8 * hi;
            out[((size_t)(b * NTOK + qrow)) * OUT_DIMX + h * VAL_DIMX + f * 16 + l16] =
                f2bf(acc[f][r] * inv[r]);
        }
}

// ---------------- host-side launch ----------------
extern "C" void kernel_launch(void* const* d_in, const int* in_sizes, int n_in,
                              void* d_out, int out_size, void* d_ws, size_t ws_size,
                              hipStream_t stream) {
    (void)in_sizes; (void)n_in; (void)out_size; (void)ws_size;
    const float* x     = (const float*)d_in[0];
    const float* gamma = (const float*)d_in[1];
    const float* beta  = (const float*)d_in[2];
    const float* Wqkv  = (const float*)d_in[3];
    const float* bqkv  = (const float*)d_in[4];
    const float* Wproj = (const float*)d_in[5];
    const float* bproj = (const float*)d_in[6];
    const float* ab    = (const float*)d_in[7];
    const int*   idxs  = (const int*)d_in[8];

    char* ws = (char*)d_ws;
    unsigned short* xn       = (unsigned short*)ws; ws += (size_t)BATCH * NTOK * DIMX * 2;
    unsigned short* wqkv_bf  = (unsigned short*)ws; ws += (size_t)QKV_OUT * DIMX * 2;
    unsigned short* wproj_bf = (unsigned short*)ws; ws += (size_t)DIMX * OUT_DIMX * 2;
    unsigned short* qkvb     = (unsigned short*)ws; ws += (size_t)BATCH * NTOK * QKV_OUT * 2;
    float*          bmx      = (float*)ws;          ws += (size_t)NUM_HEADS * NTOK * NTOK * 4;
    unsigned short* aout     = (unsigned short*)ws; ws += (size_t)BATCH * NTOK * OUT_DIMX * 2;

    cvt_kernel<<<(QKV_OUT * DIMX) / 256, 256, 0, stream>>>(Wqkv, wqkv_bf, QKV_OUT * DIMX);
    cvt_kernel<<<(DIMX * OUT_DIMX) / 256, 256, 0, stream>>>(Wproj, wproj_bf, DIMX * OUT_DIMX);
    ln_kernel<<<BATCH * NTOK, 256, 0, stream>>>(x, gamma, beta, xn);
    bias_kernel<<<(NUM_HEADS * NTOK * NTOK) / 256, 256, 0, stream>>>(ab, idxs, bmx);
    // QKV GEMM: (16384 x 512) @ (3072 x 512)^T -> bf16 qkv (Q pre-scaled by log2e/sqrt(dk))
    gemm_kernel<0, 1><<<dim3(QKV_OUT / 256, (BATCH * NTOK) / 64), 256, 0, stream>>>(
        xn, wqkv_bf, bqkv, qkvb, BATCH * NTOK, QKV_OUT, DIMX);
    // flash attention per (qtile, head, batch)
    attn_kernel<<<dim3(NTOK / 64, NUM_HEADS, BATCH), 128, 0, stream>>>(qkvb, bmx, aout);
    // projection GEMM: (16384 x 2048) @ (512 x 2048)^T -> fp32 out
    gemm_kernel<1, 0><<<dim3(DIMX / 256, (BATCH * NTOK) / 64), 256, 0, stream>>>(
        aout, wproj_bf, bproj, d_out, BATCH * NTOK, DIMX, OUT_DIMX);
}
